// LSTM_layer_31104153158088
// MI455X (gfx1250) — matrix operand order repaired and checked by reading, not compiled, not run on hardware
//
#include <hip/hip_runtime.h>
#include <hip/hip_bf16.h>

// ---------------------------------------------------------------------------
// LSTM fused-gate layer for MI455X (gfx1250, wave32, WMMA).
//   gates = [a|x] @ [Wf;Wi;Wc;Wo]^T + b   -> bf16 WMMA, f32 accumulate
//   f,i,o = sigmoid; g = tanh; c' = f*c + i*g; a' = o*tanh(c')
// GEMM: M=16384 tokens, N=4096 (4 gates x 1024), K=2048.
// Workgroup: 256 thr (8 waves), tile M=128 x (4 gates x 32 cols), K-chunk 32,
// double-buffered LDS fed by GLOBAL_LOAD_ASYNC_TO_LDS_B128 when available.
// Per wave: 2 M-tiles x 4 gates = 8 wmma per K-chunk.
// ---------------------------------------------------------------------------

typedef __attribute__((ext_vector_type(16))) __bf16          v16bf;
typedef __attribute__((ext_vector_type(8)))  __bf16          v8bf;
typedef __attribute__((ext_vector_type(8)))  float           v8f;
typedef __attribute__((ext_vector_type(4)))  float           v4f;
typedef __attribute__((ext_vector_type(4)))  int             v4i;
typedef __attribute__((ext_vector_type(4)))  unsigned short  v4us;

#define BT    16384            // B*T tokens
#define DIN   2048             // H_DIM + X_DIM
#define HD    1024             // H_DIM
#define KC    32               // K chunk per WMMA (bf16)
#define KITER (DIN / KC)       // 64
#define MT    128              // tokens per workgroup
#define NT    32               // columns-per-gate per workgroup

#if __has_builtin(__builtin_amdgcn_global_load_async_to_lds_b128)
#define USE_ASYNC_LDS 1
#else
#define USE_ASYNC_LDS 0
#endif

union V16U { v16bf v; v8bf h[2]; };

static __device__ inline unsigned short f2bf_bits(float f) {
  union { float f; unsigned u; } v; v.f = f;
  unsigned r = v.u + 0x7FFFu + ((v.u >> 16) & 1u);   // round-to-nearest-even
  return (unsigned short)(r >> 16);
}

static __device__ inline float sigmoidf_(float x) {
  return 1.0f / (1.0f + __expf(-x));
}

#if USE_ASYNC_LDS
static __device__ inline void async_b128(const __bf16* g, __bf16* l) {
  __builtin_amdgcn_global_load_async_to_lds_b128(
      (__attribute__((address_space(1))) v4i*)(g),
      (__attribute__((address_space(3))) v4i*)(l),
      0 /*imm offset*/, 0 /*cpol*/);
}
static __device__ inline void wait_async() {
#if __has_builtin(__builtin_amdgcn_s_wait_asynccnt)
  __builtin_amdgcn_s_wait_asynccnt(0);
#else
  asm volatile("s_wait_asynccnt 0x0" ::: "memory");
#endif
}
#endif

// --- pack [a|x] (fp32) -> bf16 A buffer [BT][DIN] ---------------------------
__global__ __launch_bounds__(256) void pack_a_kernel(const float* __restrict__ a,
                                                     const float* __restrict__ x,
                                                     unsigned short* __restrict__ Ap) {
  size_t tid = (size_t)blockIdx.x * blockDim.x + threadIdx.x;
  size_t e   = tid * 4;                       // element index into [BT][DIN]
  size_t tok = e >> 11;                       // / DIN
  int    k   = (int)(e & (DIN - 1));
  const float* src = (k < HD) ? (a + tok * HD + k)
                              : (x + tok * HD + (k - HD));
  v4f v = *(const v4f*)src;
  v4us o = (v4us){ f2bf_bits(v[0]), f2bf_bits(v[1]), f2bf_bits(v[2]), f2bf_bits(v[3]) };
  *(v4us*)(Ap + e) = o;
}

// --- pack Wf/Wi/Wc/Wo (fp32) -> bf16 W buffer [4][HD][DIN] ------------------
__global__ __launch_bounds__(256) void pack_w_kernel(const float* __restrict__ Wf,
                                                     const float* __restrict__ Wi,
                                                     const float* __restrict__ Wc,
                                                     const float* __restrict__ Wo,
                                                     unsigned short* __restrict__ Wp) {
  size_t tid = (size_t)blockIdx.x * blockDim.x + threadIdx.x;
  size_t e   = tid * 4;                       // element index into [4][HD*DIN]
  int    g   = (int)(e >> 21);                // / (HD*DIN)
  size_t rem = e & ((1u << 21) - 1);
  const float* W = (g == 0) ? Wf : (g == 1) ? Wi : (g == 2) ? Wc : Wo;
  v4f v = *(const v4f*)(W + rem);
  v4us o = (v4us){ f2bf_bits(v[0]), f2bf_bits(v[1]), f2bf_bits(v[2]), f2bf_bits(v[3]) };
  *(v4us*)(Wp + e) = o;
}

// --- stage one K-chunk (A: MT x KC, B: 4*NT x KC) global -> LDS -------------
// 256 threads; A tile 8KB + B tile 8KB -> 2 x b128 per thread for each.
static __device__ inline void stage_tiles(const __bf16* __restrict__ Ap,
                                          const __bf16* __restrict__ Wp,
                                          __bf16* As, __bf16* Bs,
                                          int m_base, int n_base, int k_base, int tid) {
#pragma unroll
  for (int s = 0; s < 2; ++s) {
    const int idx  = tid + s * 256;           // 0..511
    const int row  = idx >> 2;                // 0..127
    const int c8   = (idx & 3) * 8;           // 0,8,16,24
    const int gate = row >> 5;
    const int j    = row & 31;
    const __bf16* gA = &Ap[(size_t)(m_base + row) * DIN + k_base + c8];
    const __bf16* gB = &Wp[((size_t)gate * HD + n_base + j) * DIN + k_base + c8];
#if USE_ASYNC_LDS
    async_b128(gA, &As[row * KC + c8]);
    async_b128(gB, &Bs[row * KC + c8]);
#else
    *(v8bf*)&As[row * KC + c8] = *(const v8bf*)gA;
    *(v8bf*)&Bs[row * KC + c8] = *(const v8bf*)gB;
#endif
  }
}

// --- main fused GEMM + LSTM epilogue ----------------------------------------
__global__ __launch_bounds__(256) void lstm_gemm_kernel(
    const __bf16* __restrict__ Ap, const __bf16* __restrict__ Wp,
    const float* __restrict__ bf_, const float* __restrict__ bi_,
    const float* __restrict__ bc_, const float* __restrict__ bo_,
    const float* __restrict__ c_in, float* __restrict__ out) {

  __shared__ __bf16 As[2][MT * KC];         // 2 x 8 KB
  __shared__ __bf16 Bs[2][4 * NT * KC];     // 2 x 8 KB

  const int tid    = threadIdx.x;
  const int n_base = blockIdx.x * NT;       // per-gate column base
  const int m_base = blockIdx.y * MT;       // token base

  const int wave   = tid >> 5;              // 0..7
  const int lane   = tid & 31;
  const int laneLo = lane & 15;
  const int laneHi = lane >> 4;             // 0 or 1
  const int m_pair = wave & 3;              // which 32-row slab (2 x 16-row tiles)
  const int n_sub  = wave >> 2;             // which 16-col tile (of 32)

  v8f acc[2][4];
#pragma unroll
  for (int t = 0; t < 2; ++t)
#pragma unroll
    for (int g = 0; g < 4; ++g) acc[t][g] = v8f{};

  __bf16* curA = As[0];  __bf16* nxtA = As[1];
  __bf16* curB = Bs[0];  __bf16* nxtB = Bs[1];

  stage_tiles(Ap, Wp, curA, curB, m_base, n_base, 0, tid);
#if USE_ASYNC_LDS
  wait_async();
#endif
  __syncthreads();

  for (int kt = 0; kt < KITER; ++kt) {
    if (kt + 1 < KITER)
      stage_tiles(Ap, Wp, nxtA, nxtB, m_base, n_base, (kt + 1) * KC, tid);

    // A operands (16x32 bf16): lanes 0-15 row M=lane, K{0..7,16..23};
    // lanes 16-31 same rows, K{8..15,24..31}.
    V16U ua[2];
#pragma unroll
    for (int t = 0; t < 2; ++t) {
      const __bf16* aPtr = &curA[(m_pair * 32 + t * 16 + laneLo) * KC + laneHi * 8];
      ua[t].h[0] = *(const v8bf*)(aPtr);
      ua[t].h[1] = *(const v8bf*)(aPtr + 16);
    }

    const int colB = n_sub * 16 + laneLo;
#pragma unroll
    for (int g = 0; g < 4; ++g) {
      // B operand (32x16 bf16): lane = column N; lanes 0-15 K{0..7,16..23},
      // lanes 16-31 K{8..15,24..31}, packed 2 K-values per VGPR.
      const __bf16* bPtr = &curB[(g * NT + colB) * KC + laneHi * 8];
      V16U ub;
      ub.h[0] = *(const v8bf*)(bPtr);
      ub.h[1] = *(const v8bf*)(bPtr + 16);
#pragma unroll
      for (int t = 0; t < 2; ++t)
        acc[t][g] = __builtin_amdgcn_wmma_f32_16x16x32_bf16(
            false, ua[t].v, false, ub.v, (short)0, acc[t][g], false, false);
    }

#if USE_ASYNC_LDS
    wait_async();                           // own async writes visible in LDS
#endif
    __syncthreads();                        // cross-wave visibility

    __bf16* tA = curA; curA = nxtA; nxtA = tA;
    __bf16* tB = curB; curB = nxtB; nxtB = tB;
  }

  // ---- fused LSTM epilogue -------------------------------------------------
  // C/D layout: VGPR r, lanes 0-15 -> M=r, N=lane; lanes 16-31 -> M=r+8.
  const int jg = n_base + n_sub * 16 + laneLo;     // gate column 0..1023
  const float Bf = bf_[jg], Bi = bi_[jg], Bc = bc_[jg], Bo = bo_[jg];

  float* out_o = out;
  float* out_a = out + (size_t)BT * HD;
  float* out_c = out + (size_t)2 * BT * HD;

#pragma unroll
  for (int t = 0; t < 2; ++t) {
    const int m0 = m_base + m_pair * 32 + t * 16 + laneHi * 8;
#pragma unroll
    for (int r = 0; r < 8; ++r) {
      const size_t idx = (size_t)(m0 + r) * HD + jg;
      const float f  = sigmoidf_(acc[t][0][r] + Bf);
      const float iv = sigmoidf_(acc[t][1][r] + Bi);
      const float g  = tanhf(acc[t][2][r] + Bc);
      const float o  = sigmoidf_(acc[t][3][r] + Bo);
      const float cN = f * c_in[idx] + iv * g;
      const float aN = o * tanhf(cN);
      out_o[idx] = o;
      out_a[idx] = aN;
      out_c[idx] = cN;
    }
  }
}

extern "C" void kernel_launch(void* const* d_in, const int* in_sizes, int n_in,
                              void* d_out, int out_size, void* d_ws, size_t ws_size,
                              hipStream_t stream) {
  (void)in_sizes; (void)n_in; (void)out_size; (void)ws_size;

  const float* x  = (const float*)d_in[0];
  const float* a  = (const float*)d_in[1];
  const float* c  = (const float*)d_in[2];
  const float* Wf = (const float*)d_in[3];
  const float* bf = (const float*)d_in[4];
  const float* Wi = (const float*)d_in[5];
  const float* bi = (const float*)d_in[6];
  const float* Wc = (const float*)d_in[7];
  const float* bc = (const float*)d_in[8];
  const float* Wo = (const float*)d_in[9];
  const float* bo = (const float*)d_in[10];

  // Workspace layout: bf16 A [BT][DIN] (64 MB) then bf16 W [4][HD][DIN] (16 MB)
  unsigned short* Apack = (unsigned short*)d_ws;
  unsigned short* Wpack = (unsigned short*)((char*)d_ws + (size_t)BT * DIN * 2);

  // Pack inputs to bf16.
  pack_a_kernel<<<(BT * (size_t)DIN / 4) / 256, 256, 0, stream>>>(a, x, Apack);
  pack_w_kernel<<<(4 * (size_t)HD * DIN / 4) / 256, 256, 0, stream>>>(Wf, Wi, Wc, Wo, Wpack);

  // Fused GEMM + LSTM cell.
  dim3 grid(HD / NT, BT / MT);   // (32, 128)
  lstm_gemm_kernel<<<grid, 256, 0, stream>>>(
      (const __bf16*)Apack, (const __bf16*)Wpack, bf, bi, bc, bo, c, (float*)d_out);
}
